// DocREModel_16930761081450
// MI455X (gfx1250) — matrix operand order repaired and checked by reading, not compiled, not run on hardware
//
#include <hip/hip_runtime.h>
#include <hip/hip_bf16.h>

typedef __attribute__((ext_vector_type(16))) _Float16 v16h;
typedef __attribute__((ext_vector_type(8)))  _Float16 v8h;
typedef __attribute__((ext_vector_type(8)))  float    v8f;

#define B_   4
#define L_   1024
#define H_   768
#define HEADS_ 12
#define E_   22
#define MPD_ 128
#define EMB_ 512
#define M_   (B_*MPD_)   // 512 mentions
#define NE_  (B_*E_)     // 88 entities
#define EPAD 32          // padded entities per batch (WMMA M tiles)

// union to build WMMA fragments straight from two 128-bit loads (no repack movs)
union frag16 { v16h v; struct { v8h lo, hi; } p; };

// ---------------- fused f32 -> f16 transpose (out[c][r] = (h)in[r][c]) ----------------
// block (32,8); grid (C/32, R/32, nbatch). in: R x C row-major, out: C x R row-major.
__global__ void transpose_f16_kernel(const float* __restrict__ in,
                                     _Float16* __restrict__ out,
                                     int R, int C) {
    __shared__ _Float16 tile[32][33];
    const size_t zin  = (size_t)blockIdx.z * R * C;
    const size_t zout = (size_t)blockIdx.z * R * C;
    const int c = blockIdx.x * 32 + threadIdx.x;
    const int r0 = blockIdx.y * 32;
    #pragma unroll
    for (int j = 0; j < 4; ++j) {
        int r = r0 + threadIdx.y + j * 8;
        tile[threadIdx.y + j * 8][threadIdx.x] = (_Float16)in[zin + (size_t)r * C + c];
    }
    __syncthreads();
    const int rr = r0 + threadIdx.x;
    #pragma unroll
    for (int j = 0; j < 4; ++j) {
        int cc = blockIdx.x * 32 + threadIdx.y + j * 8;
        out[zout + (size_t)cc * R + rr] = tile[threadIdx.x][threadIdx.y + j * 8];
    }
}

// ---------------- entity logsumexp embedding ----------------
// grid = NE_ (88), block = 256. entity_emb[e*H + h] = logsumexp over mentions.
__global__ void entity_emb_kernel(const float* __restrict__ seq,
                                  const int* __restrict__ m_batch,
                                  const int* __restrict__ m_pos,
                                  const int* __restrict__ m_ent,
                                  float* __restrict__ entity_emb) {
    __shared__ int list[M_];
    __shared__ int cnt;
    const int e = blockIdx.x;
    const int tid = threadIdx.x;
    if (tid == 0) cnt = 0;
    __syncthreads();
    for (int m = tid; m < M_; m += 256) {
        if (m_ent[m] == e) { int idx = atomicAdd(&cnt, 1); list[idx] = m; }
    }
    __syncthreads();
    const int c = cnt;
    for (int h = tid; h < H_; h += 256) {
        if (c == 0) { entity_emb[(size_t)e * H_ + h] = 0.0f; continue; }
        float mx = -3.402823466e38f;
        for (int i = 0; i < c; ++i) {
            int m = list[i];
            float v = seq[((size_t)m_batch[m] * L_ + m_pos[m]) * H_ + h];
            mx = fmaxf(mx, v);
        }
        float s = 0.0f;
        for (int i = 0; i < c; ++i) {
            int m = list[i];
            float v = seq[((size_t)m_batch[m] * L_ + m_pos[m]) * H_ + h];
            s += expf(v - mx);
        }
        entity_emb[(size_t)e * H_ + h] = logf(s + 1e-30f) + mx;
    }
}

// ---------------- entity attention (normalized), f16, padded rows ----------------
// grid = B_*EPAD (128), block = 256. eatt_h[(b*EPAD+le)*L + l]; rows le>=E_ are zero.
__global__ void eatt_kernel(const float* __restrict__ att,
                            const int* __restrict__ m_batch,
                            const int* __restrict__ m_pos,
                            const int* __restrict__ m_ent,
                            _Float16* __restrict__ eatt_h) {
    __shared__ int list[M_];
    __shared__ int cnt;
    __shared__ float red[256];
    const int pe = blockIdx.x;
    const int b = pe >> 5, le = pe & 31;
    const int tid = threadIdx.x;
    _Float16* out = eatt_h + (size_t)pe * L_;
    if (le >= E_) {                       // zero pad rows 22..31
        for (int l = tid; l < L_; l += 256) out[l] = (_Float16)0.0f;
        return;
    }
    const int e = b * E_ + le;
    if (tid == 0) cnt = 0;
    __syncthreads();
    for (int m = tid; m < M_; m += 256) {
        if (m_ent[m] == e) { int idx = atomicAdd(&cnt, 1); list[idx] = m; }
    }
    __syncthreads();
    const int c = cnt;
    float acc[4] = {0.f, 0.f, 0.f, 0.f};
    for (int i = 0; i < c; ++i) {
        int m = list[i];
        const float* base = att + ((size_t)m_batch[m] * HEADS_) * L_ * L_
                                + (size_t)m_pos[m] * L_;
        for (int head = 0; head < HEADS_; ++head) {
            const float* row = base + (size_t)head * L_ * L_;
            #pragma unroll
            for (int j = 0; j < 4; ++j) acc[j] += row[tid + j * 256];
        }
    }
    const float inv = 1.0f / (((float)c + 1e-5f) * (float)HEADS_);
    float val[4], part = 0.0f;
    #pragma unroll
    for (int j = 0; j < 4; ++j) { val[j] = acc[j] * inv; part += val[j]; }
    red[tid] = part;
    __syncthreads();
    for (int s = 128; s > 0; s >>= 1) {
        if (tid < s) red[tid] += red[tid + s];
        __syncthreads();
    }
    const float norm = 1.0f / (red[0] + 1e-5f);
    #pragma unroll
    for (int j = 0; j < 4; ++j) out[tid + j * 256] = (_Float16)(val[j] * norm);
}

// ---------------- GEMM1: e_context = e_att @ seq, + entity_emb -> e_cg (f16) ----------
// grid = B_*2*48 waves, block = 32 (one wave, EXEC all ones around WMMA).
// B operand from transposed copy seqT_h[b][h][l] -> contiguous 16B loads.
__global__ void __launch_bounds__(32)
gemm1_kernel(const _Float16* __restrict__ eatt_h,
             const _Float16* __restrict__ seqT_h,
             const float* __restrict__ entity_emb,
             _Float16* __restrict__ ecg_h) {
    const int wid = blockIdx.x;
    const int b  = wid / 96;
    const int rem = wid % 96;
    const int mt = rem / 48, nt = rem % 48;
    const int m0 = mt * 16, n0 = nt * 16;
    const int lane = threadIdx.x;
    const int row = lane & 15, hs = lane >> 4;
    const _Float16* aRow = eatt_h + (size_t)(b * EPAD + m0 + row) * L_;
    const int col = n0 + row;            // B/N column (h) for this lane
    const _Float16* bRow = seqT_h + ((size_t)b * H_ + col) * L_;
    v8f c = {};
    #pragma unroll 4
    for (int k0 = 0; k0 < L_; k0 += 32) {
        frag16 a, bf;
        a.p.lo  = *(const v8h*)(aRow + k0 + hs * 8);
        a.p.hi  = *(const v8h*)(aRow + k0 + 16 + hs * 8);
        bf.p.lo = *(const v8h*)(bRow + k0 + hs * 8);
        bf.p.hi = *(const v8h*)(bRow + k0 + 16 + hs * 8);
        c = __builtin_amdgcn_wmma_f32_16x16x32_f16(false, a.v, false, bf.v,
                                                   (short)0, c, false, false);
    }
    #pragma unroll
    for (int r = 0; r < 8; ++r) {
        const int e = m0 + r + hs * 8;   // padded entity row within batch
        const int h = col;
        _Float16 v = (_Float16)0.0f;
        if (e < E_)
            v = (_Float16)(c[r] + entity_emb[((size_t)b * E_ + e) * H_ + h]);
        ecg_h[((size_t)b * EPAD + e) * H_ + h] = v;
    }
}

// ---------------- GEMM2: out = tanh(e_cg @ w_trans + b_trans) ----------------
// grid = 8*32 waves, block = 32. Padded M = 128 rows. B from wT_h[j][h].
__global__ void __launch_bounds__(32)
gemm2_kernel(const _Float16* __restrict__ ecg_h,
             const _Float16* __restrict__ wT_h,
             const float* __restrict__ b_trans,
             float* __restrict__ out) {
    const int wid = blockIdx.x;
    const int mt = wid / 32, nt = wid % 32;
    const int m0 = mt * 16, n0 = nt * 16;
    const int lane = threadIdx.x;
    const int row = lane & 15, hs = lane >> 4;
    const _Float16* aRow = ecg_h + (size_t)(m0 + row) * H_;
    const int col = n0 + row;            // output column j
    const _Float16* bRow = wT_h + (size_t)col * H_;
    v8f c = {};
    #pragma unroll 4
    for (int k0 = 0; k0 < H_; k0 += 32) {
        frag16 a, bf;
        a.p.lo  = *(const v8h*)(aRow + k0 + hs * 8);
        a.p.hi  = *(const v8h*)(aRow + k0 + 16 + hs * 8);
        bf.p.lo = *(const v8h*)(bRow + k0 + hs * 8);
        bf.p.hi = *(const v8h*)(bRow + k0 + 16 + hs * 8);
        c = __builtin_amdgcn_wmma_f32_16x16x32_f16(false, a.v, false, bf.v,
                                                   (short)0, c, false, false);
    }
    #pragma unroll
    for (int r = 0; r < 8; ++r) {
        const int pr = m0 + r + hs * 8;      // padded global row 0..127
        const int bb = pr >> 5, le = pr & 31;
        const int j = col;
        if (le < E_)
            out[((size_t)bb * E_ + le) * EMB_ + j] = tanhf(c[r] + b_trans[j]);
    }
}

// ---------------- launch ----------------
extern "C" void kernel_launch(void* const* d_in, const int* in_sizes, int n_in,
                              void* d_out, int out_size, void* d_ws, size_t ws_size,
                              hipStream_t stream) {
    const float* seq  = (const float*)d_in[0];   // (4,1024,768)
    const float* att  = (const float*)d_in[1];   // (4,12,1024,1024)
    const float* wtr  = (const float*)d_in[2];   // (768,512)
    const float* btr  = (const float*)d_in[3];   // (512,)
    const int* m_batch = (const int*)d_in[4];    // (512,)
    const int* m_pos   = (const int*)d_in[5];    // (512,)
    const int* m_ent   = (const int*)d_in[6];    // (512,)
    float* out = (float*)d_out;                  // (4,22,512)

    char* ws = (char*)d_ws;
    // all offsets multiples of 256 bytes
    float*    entity_emb = (float*)(ws + 0);             //  88*768*4 = 270336
    _Float16* eatt_h     = (_Float16*)(ws + 270336);     // 128*1024*2 = 262144
    _Float16* ecg_h      = (_Float16*)(ws + 532480);     // 128*768*2  = 196608
    _Float16* seqT_h     = (_Float16*)(ws + 729088);     // 3145728*2  = 6291456 (b,H,L)
    _Float16* wT_h       = (_Float16*)(ws + 7020544);    // 393216*2   = 786432  (EMB,H)

    // fused convert + transpose: seq (L x H per batch) -> seqT (H x L per batch)
    {
        dim3 blk(32, 8, 1);
        dim3 grdS(H_ / 32, L_ / 32, B_);     // C/32, R/32, batches
        transpose_f16_kernel<<<grdS, blk, 0, stream>>>(seq, seqT_h, L_, H_);
        dim3 grdW(EMB_ / 32, H_ / 32, 1);
        transpose_f16_kernel<<<grdW, blk, 0, stream>>>(wtr, wT_h, H_, EMB_);
    }

    entity_emb_kernel<<<NE_, 256, 0, stream>>>(seq, m_batch, m_pos, m_ent, entity_emb);
    eatt_kernel<<<B_ * EPAD, 256, 0, stream>>>(att, m_batch, m_pos, m_ent, eatt_h);

    gemm1_kernel<<<B_ * 2 * (H_ / 16), 32, 0, stream>>>(eatt_h, seqT_h, entity_emb, ecg_h);
    gemm2_kernel<<<(B_ * EPAD / 16) * (EMB_ / 16), 32, 0, stream>>>(ecg_h, wT_h, btr, out);
}
